// TranslationIPA_13520557048388
// MI455X (gfx1250) — matrix operand order repaired and checked by reading, not compiled, not run on hardware
//
#include <hip/hip_runtime.h>
#include <hip/hip_bf16.h>
#include <math.h>

// Problem constants (B=1)
#define Nres 512
#define CSd  384
#define CZd  128
#define Hh   8
#define Cc   256
#define PQp  8
#define PVp  12
#define KCAT 288            // 256 (channels) + 24 (points) padded to mult of 32
#define NM   (Nres * Nres)  // 262144
#define FEAT 2688           // H*(C + 4*PV + CZ/4)

typedef __attribute__((ext_vector_type(16))) __bf16 v16bf;
typedef __attribute__((ext_vector_type(2)))  __bf16 bf16x2;
typedef __attribute__((ext_vector_type(8)))  float  v8f;
typedef unsigned int u32;
typedef __attribute__((ext_vector_type(4))) unsigned int u32x4;
typedef __attribute__((ext_vector_type(8))) unsigned int u32x8;

__device__ __forceinline__ int lane_id() { return (int)(threadIdx.x & 31u); }

__device__ __forceinline__ float softplusf(float x) {
  return (x > 20.f) ? x : log1pf(__expf(x));
}

__device__ __forceinline__ u32 pack_bf16(float a, float b) {
  bf16x2 t;
  t[0] = (__bf16)a;
  t[1] = (__bf16)b;
  return __builtin_bit_cast(u32, t);
}

__device__ __forceinline__ v8f wmma_bf16(v16bf a, v16bf b, v8f c) {
  return __builtin_amdgcn_wmma_f32_16x16x32_bf16(false, a, false, b, (short)0, c,
                                                 false, false);
}

// LDS byte offset of a generic pointer known to point into LDS.
__device__ __forceinline__ u32 lds_offset(const void* p) {
  return (u32)(size_t)(__attribute__((address_space(3))) const void*)p;
}

// Async DMA: 16B global -> LDS, tracked by ASYNCcnt (CDNA5).
__device__ __forceinline__ void async_copy_b128(u32 lds_off, const void* gptr) {
  asm volatile("global_load_async_to_lds_b128 %0, %1, off"
               :
               : "v"(lds_off), "v"((unsigned long long)(size_t)gptr)
               : "memory");
}
__device__ __forceinline__ void async_wait0() {
  asm volatile("s_wait_asynccnt 0x0" ::: "memory");
}

// A fragment (16x32 bf16): rows m = lane&15; per ISA 7.12.2 the 8 VGPRs cover
// K = khalf+{0..7} (v0..v3, 16 contiguous bytes) and +16 (v4..v7).
// Requires strideM % 8 == 0 and 16B-aligned base.
__device__ __forceinline__ v16bf load_a_frag(const __bf16* base, int strideM, int k0) {
  const int l = lane_id();
  const int m = l & 15;
  const int khalf = (l < 16) ? 0 : 8;
  const u32x4* p = (const u32x4*)(base + (size_t)m * strideM + k0 + khalf);
  u32x4 lo = p[0];
  u32x4 hi = p[2];  // byte offset +32
  u32x8 w;
  w[0] = lo[0]; w[1] = lo[1]; w[2] = lo[2]; w[3] = lo[3];
  w[4] = hi[0]; w[5] = hi[1]; w[6] = hi[2]; w[7] = hi[3];
  return __builtin_bit_cast(v16bf, w);
}

// --- Fragment-swizzled B operand storage -----------------------------------
// A 32(K) x 16(N) fragment is stored as [frag][lane][8] dwords so each lane's
// 8 operand dwords are contiguous (2x b128 loads). Element (k,n) lives at
// u32 index bsw_idx_u32(k,n,NT) with bf16 parity (k&1); lane = (n&15) plus 16
// when (k&8); dword v = ((k&7)>>1) + 4*((k>>4)&1).
__device__ __forceinline__ size_t bsw_idx_u32(int k, int n, int NT) {
  int l = (n & 15) | (((k >> 3) & 1) << 4);
  int frag = (k >> 5) * NT + (n >> 4);
  return ((size_t)frag * 32 + l) * 8 + (((k >> 1) & 3) | (((k >> 4) & 1) << 2));
}

__device__ __forceinline__ v16bf load_b_frag_sw(const u32* base, int frag) {
  const int l = lane_id();
  const u32x4* p = (const u32x4*)(base + ((size_t)frag * 32 + l) * 8);
  u32x4 a = p[0];
  u32x4 b = p[1];
  u32x8 w;
  w[0] = a[0]; w[1] = a[1]; w[2] = a[2]; w[3] = a[3];
  w[4] = b[0]; w[5] = b[1]; w[6] = b[2]; w[7] = b[3];
  return __builtin_bit_cast(v16bf, w);
}

// ---------------------------------------------------------------------------
// Generic C[M,N] = A[M,K] @ W[K,N] + bias, f32 in/out, bf16 WMMA compute.
// M % 64 == 0, K % 32 == 0, N guarded.
// ---------------------------------------------------------------------------
__global__ __launch_bounds__(128) void gemm_f32_bf16(
    const float* __restrict__ A, const float* __restrict__ W,
    const float* __restrict__ bias, float* __restrict__ Cout, int M, int Nc, int K) {
  __shared__ __align__(16) __bf16 As[64][40];
  __shared__ __align__(16) u32 Wsw[4 * 32 * 8];  // 4 frags (32K x 64N), swizzled
  const int tid  = threadIdx.x;
  const int wave = tid >> 5;
  const int m0 = blockIdx.y * 64;
  const int n0 = blockIdx.x * 64;
  (void)M;
  v8f acc[4] = {};
  for (int kt = 0; kt < K; kt += 32) {
    // A tile: convert pairs (one v_cvt_pk_bf16_f32 + b32 store per 2 elems)
    for (int idx = tid; idx < 64 * 16; idx += 128) {
      int r = idx >> 4, cp = idx & 15;
      const float2 f = *(const float2*)&A[(size_t)(m0 + r) * K + kt + cp * 2];
      *(u32*)&As[r][cp * 2] = pack_bf16(f.x, f.y);
    }
    // W tile into swizzled-fragment layout: pack K-parity pairs into dwords
    for (int idx = tid; idx < 16 * 64; idx += 128) {
      int rp = idx >> 6, c = idx & 63;
      int col = n0 + c;
      float f0 = 0.f, f1 = 0.f;
      if (col < Nc) {
        f0 = W[(size_t)(kt + 2 * rp) * Nc + col];
        f1 = W[(size_t)(kt + 2 * rp + 1) * Nc + col];
      }
      Wsw[bsw_idx_u32(2 * rp, c, 4)] = pack_bf16(f0, f1);
    }
    if (kt + 32 < K) {
      __builtin_prefetch(&A[(size_t)(m0 + (tid & 63)) * K + kt + 32], 0, 1);
      __builtin_prefetch(&W[(size_t)(kt + 32 + (tid & 31)) * Nc + n0], 0, 1);
    }
    __syncthreads();
    v16bf af = load_a_frag(&As[0][0] + (size_t)wave * 16 * 40, 40, 0);
#pragma unroll
    for (int n = 0; n < 4; ++n) {
      v16bf bf = load_b_frag_sw(Wsw, n);
      acc[n] = wmma_bf16(af, bf, acc[n]);
    }
    __syncthreads();
  }
  const int l = lane_id();
  const int nlo = l & 15;
  const int mhi = (l < 16) ? 0 : 8;
#pragma unroll
  for (int n = 0; n < 4; ++n) {
    int col = n0 + n * 16 + nlo;
    if (col >= Nc) continue;
    float bv = bias ? bias[col] : 0.0f;
#pragma unroll
    for (int v = 0; v < 8; ++v) {
      int row = m0 + wave * 16 + mhi + v;
      Cout[(size_t)row * Nc + col] = acc[n][v] + bv;
    }
  }
}

// ---------------------------------------------------------------------------
// Fused z projections: one pass over z (512*512 x 128) -> b_bias (scaled by
// sqrt(1/3), head-major) and pair_z (bf16, fragment-swizzled over K=j).
// ---------------------------------------------------------------------------
__global__ __launch_bounds__(128) void zproj_kernel(
    const float* __restrict__ z, const float* __restrict__ wb,
    const float* __restrict__ bb, const float* __restrict__ wz,
    const float* __restrict__ bz, float* __restrict__ bbias,
    __bf16* __restrict__ pairz) {
  __shared__ __align__(16) __bf16 Zs[64][40];
  __shared__ __align__(16) u32 Wcsw[12 * 32 * 8];  // 4 ktiles x 3 ntiles, swizzled
  const int tid  = threadIdx.x;
  const int wave = tid >> 5;
  const int m0 = blockIdx.x * 64;
  for (int idx = tid; idx < 64 * 48; idx += 128) {
    int kp = idx / 48, c = idx % 48;
    float f0 = 0.f, f1 = 0.f;
    if (c < 8) {
      f0 = wb[(2 * kp) * 8 + c];
      f1 = wb[(2 * kp + 1) * 8 + c];
    } else if (c < 40) {
      f0 = wz[(2 * kp) * 32 + (c - 8)];
      f1 = wz[(2 * kp + 1) * 32 + (c - 8)];
    }
    Wcsw[bsw_idx_u32(2 * kp, c, 3)] = pack_bf16(f0, f1);
  }
  __syncthreads();
  v8f acc[3] = {};
  for (int kt = 0; kt < 128; kt += 32) {
    for (int idx = tid; idx < 64 * 16; idx += 128) {
      int r = idx >> 4, cp = idx & 15;
      const float2 f = *(const float2*)&z[(size_t)(m0 + r) * 128 + kt + cp * 2];
      *(u32*)&Zs[r][cp * 2] = pack_bf16(f.x, f.y);
    }
    __syncthreads();
    v16bf af = load_a_frag(&Zs[0][0] + (size_t)wave * 16 * 40, 40, 0);
#pragma unroll
    for (int n = 0; n < 3; ++n) {
      v16bf bf = load_b_frag_sw(Wcsw, (kt >> 5) * 3 + n);
      acc[n] = wmma_bf16(af, bf, acc[n]);
    }
    __syncthreads();
  }
  const int l = lane_id();
  const int nlo = l & 15;
  const int mhi = (l < 16) ? 0 : 8;
#pragma unroll
  for (int n = 0; n < 3; ++n) {
    int col = n * 16 + nlo;
#pragma unroll
    for (int v = 0; v < 8; ++v) {
      int m = m0 + wave * 16 + mhi + v;
      float val = acc[n][v];
      if (col < 8) {
        bbias[(size_t)col * NM + m] = (val + bb[col]) * 0.57735026919f; // sqrt(1/3)
      } else if (col < 40) {
        int i = m >> 9, j = m & 511, d = col - 8;
        // swizzled B layout over K=j, NT=2 (32 cols), 8192 dwords per i
        size_t u = (size_t)i * 8192 + bsw_idx_u32(j, d, 2);
        pairz[u * 2 + (j & 1)] = (__bf16)(val + bz[d]);
      }
    }
  }
}

// ---------------------------------------------------------------------------
// Rigid transform of points + build bf16 operands:
//   qcat  [h][i][KCAT]     (A operand, row-major)
//   kcatT [h] swizzled B   (K=288 x N=512, NT=32, 73728 dwords/head)
//   vbf   [h] swizzled B   (K=512 x N=256, NT=16, 65536 dwords/head)
//   vptbf [h] swizzled B   (K=512 x N=48,  NT=3,  12288 dwords/head)
// plus Qsq/Ksq corrections. One block per residue i.
// ---------------------------------------------------------------------------
__global__ __launch_bounds__(256) void pack_kernel(
    const float* __restrict__ qbuf, const float* __restrict__ kvbuf,
    const float* __restrict__ qpbuf, const float* __restrict__ kvpbuf,
    const float* __restrict__ rot, const float* __restrict__ trans,
    const float* __restrict__ head_w, __bf16* __restrict__ qcat,
    __bf16* __restrict__ kcatT, __bf16* __restrict__ vbf,
    __bf16* __restrict__ vptbf, float* __restrict__ Qsq, float* __restrict__ Ksq) {
  const int i = blockIdx.x;
  const int tid = threadIdx.x;
  __shared__ float sqp[64][3];
  __shared__ float skv[160][3];
  __shared__ float R[9], T[3];
  if (tid < 9) R[tid] = rot[i * 9 + tid];
  if (tid < 3) T[tid] = trans[i * 3 + tid];
  __syncthreads();
  for (int p = tid; p < 64; p += 256) {
    float px = qpbuf[i * 192 + p], py = qpbuf[i * 192 + 64 + p],
          pz = qpbuf[i * 192 + 128 + p];
    sqp[p][0] = R[0] * px + R[1] * py + R[2] * pz + T[0];
    sqp[p][1] = R[3] * px + R[4] * py + R[5] * pz + T[1];
    sqp[p][2] = R[6] * px + R[7] * py + R[8] * pz + T[2];
  }
  for (int p = tid; p < 160; p += 256) {
    float px = kvpbuf[i * 480 + p], py = kvpbuf[i * 480 + 160 + p],
          pz = kvpbuf[i * 480 + 320 + p];
    skv[p][0] = R[0] * px + R[1] * py + R[2] * pz + T[0];
    skv[p][1] = R[3] * px + R[4] * py + R[5] * pz + T[1];
    skv[p][2] = R[6] * px + R[7] * py + R[8] * pz + T[2];
  }
  __syncthreads();
  if (tid < 8) {
    int h = tid;
    float qs = 0.f, ks = 0.f;
    for (int t = 0; t < 8; ++t) {
      const float* q = sqp[h * 8 + t];
      const float* k = skv[h * 20 + t];
      qs += q[0] * q[0] + q[1] * q[1] + q[2] * q[2];
      ks += k[0] * k[0] + k[1] * k[1] + k[2] * k[2];
    }
    Qsq[i * 8 + h] = qs;
    Ksq[i * 8 + h] = ks;
  }
  const float qscale = 0.03608439182435161f; // sqrt(1/(3*C))
  // qcat: write dword pairs (contiguous along c)
  for (int idx = tid; idx < 8 * (KCAT / 2); idx += 256) {
    int h = idx / (KCAT / 2), cp = idx % (KCAT / 2);
    float hw = softplusf(head_w[h]) * 0.09622504486493764f; // sqrt(1/108)
    float v01[2];
#pragma unroll
    for (int u = 0; u < 2; ++u) {
      int c = cp * 2 + u;
      float val = 0.f;
      if (c < 256)      val = qbuf[(size_t)i * 2048 + h * 256 + c] * qscale;
      else if (c < 280) { int pp = c - 256; val = sqp[h * 8 + pp / 3][pp % 3] * hw; }
      v01[u] = val;
    }
    ((u32*)qcat)[((size_t)h * Nres + i) * (KCAT / 2) + cp] =
        pack_bf16(v01[0], v01[1]);
  }
  // kcatT: swizzled B, column n = i, one dword per kpair
  for (int idx = tid; idx < 8 * (KCAT / 2); idx += 256) {
    int h = idx / (KCAT / 2), kp = idx % (KCAT / 2);
    float v01[2];
#pragma unroll
    for (int u = 0; u < 2; ++u) {
      int k = kp * 2 + u;
      float val = 0.f;
      if (k < 256)      val = kvbuf[(size_t)i * 4096 + h * 512 + k];
      else if (k < 280) { int pp = k - 256; val = skv[h * 20 + pp / 3][pp % 3]; }
      v01[u] = val;
    }
    ((u32*)kcatT)[(size_t)h * 73728 + bsw_idx_u32(kp * 2, i, 32)] =
        pack_bf16(v01[0], v01[1]);
  }
  // vbf: swizzled B over K=i (parity i&1), N=256
  for (int idx = tid; idx < 8 * 256; idx += 256) {
    int h = idx >> 8, c = idx & 255;
    size_t u = (size_t)h * 65536 + bsw_idx_u32(i, c, 16);
    vbf[u * 2 + (i & 1)] =
        (__bf16)kvbuf[(size_t)i * 4096 + h * 512 + 256 + c];
  }
  // vptbf: swizzled B over K=i, N=48
  for (int idx = tid; idx < 8 * 48; idx += 256) {
    int h = idx / 48, c = idx % 48;
    float val = 0.f;
    if (c < 36) val = skv[h * 20 + 8 + c / 3][c % 3];
    size_t u = (size_t)h * 12288 + bsw_idx_u32(i, c, 3);
    vptbf[u * 2 + (i & 1)] = (__bf16)val;
  }
}

// ---------------------------------------------------------------------------
// Fused logits (WMMA over K=288) + corrections + softmax, writes bf16 probs.
// Block: 128 threads handles 16 rows (one head). qcat tile staged to LDS via
// CDNA5 async DMA (global_load_async_to_lds_b128, ASYNCcnt).
// ---------------------------------------------------------------------------
__global__ __launch_bounds__(128) void logits_softmax_kernel(
    const __bf16* __restrict__ qcat, const __bf16* __restrict__ kcatT,
    const float* __restrict__ bbias, const float* __restrict__ Qsq,
    const float* __restrict__ Ksq, const float* __restrict__ mask,
    const float* __restrict__ head_w, __bf16* __restrict__ a_out) {
  const int h  = blockIdx.y;
  const int i0 = blockIdx.x * 16;
  const int tid  = threadIdx.x;
  const int wave = tid >> 5;
  __shared__ __align__(16) __bf16 qs[16][KCAT];
  __shared__ float lg[16][512];
  // Async-stage 16 x 288 bf16 (9216 B) as 576 x b128 DMA transfers.
  for (int idx = tid; idx < 16 * (KCAT / 8); idx += 128) {
    int r = idx / (KCAT / 8), c8 = idx % (KCAT / 8);
    const void* g = &qcat[((size_t)h * Nres + i0 + r) * KCAT + c8 * 8];
    async_copy_b128(lds_offset(&qs[r][c8 * 8]), g);
  }
  async_wait0();
  __syncthreads();
  const float hw = softplusf(head_w[h]) * 0.09622504486493764f;
  const int l = lane_id(), nlo = l & 15, mhi = (l < 16) ? 0 : 8;
  const int jbase = wave * 128;
  const u32* kbase = (const u32*)kcatT + (size_t)h * 73728;
#pragma unroll
  for (int t = 0; t < 8; ++t) {
    v8f acc = {};
    const int jt = (jbase + t * 16) >> 4;
    for (int kt = 0; kt < KCAT; kt += 32) {
      v16bf af = load_a_frag(&qs[0][0], KCAT, kt);
      v16bf bf = load_b_frag_sw(kbase, (kt >> 5) * 32 + jt);
      acc = wmma_bf16(af, bf, acc);
    }
    int j = jbase + t * 16 + nlo;
    float mj  = mask[j];
    float ksj = Ksq[j * 8 + h];
#pragma unroll
    for (int v = 0; v < 8; ++v) {
      int m = mhi + v;
      int i = i0 + m;
      float val = acc[v] + bbias[(size_t)h * NM + (size_t)i * 512 + j] -
                  0.5f * hw * (Qsq[i * 8 + h] + ksj) +
                  100000.0f * (mask[i] * mj - 1.0f);
      lg[m][j] = val;
    }
  }
  __syncthreads();
  for (int rr = 0; rr < 4; ++rr) {
    int m = wave + rr * 4;
    float mx = -3.0e38f;
    for (int j = l; j < 512; j += 32) mx = fmaxf(mx, lg[m][j]);
#pragma unroll
    for (int off = 16; off > 0; off >>= 1) mx = fmaxf(mx, __shfl_xor(mx, off, 32));
    float sum = 0.f;
    for (int j = l; j < 512; j += 32) sum += __expf(lg[m][j] - mx);
#pragma unroll
    for (int off = 16; off > 0; off >>= 1) sum += __shfl_xor(sum, off, 32);
    float inv = 1.0f / sum;
    for (int j = l; j < 512; j += 32)
      a_out[((size_t)h * Nres + i0 + m) * Nres + j] =
          (__bf16)(__expf(lg[m][j] - mx) * inv);
  }
}

// ---------------------------------------------------------------------------
// o = a@v and o_pt = a@v_pts per head (concat cols: 256 + 48).
// ---------------------------------------------------------------------------
__global__ __launch_bounds__(128) void av_kernel(
    const __bf16* __restrict__ a_bf, const __bf16* __restrict__ vbf,
    const __bf16* __restrict__ vptbf, float* __restrict__ feats,
    float* __restrict__ optg) {
  const int h  = blockIdx.y;
  const int m0 = blockIdx.x * 64;
  const int tid = threadIdx.x, wave = tid >> 5;
  const int l = lane_id(), nlo = l & 15, mhi = (l < 16) ? 0 : 8;
  const __bf16* abase = a_bf + ((size_t)h * Nres + m0 + wave * 16) * Nres;
  const u32* vb  = (const u32*)vbf + (size_t)h * 65536;
  const u32* vpb = (const u32*)vptbf + (size_t)h * 12288;
  for (int ct = 0; ct < 19; ++ct) {
    int col0 = ct * 16;
    v8f acc = {};
    for (int kt = 0; kt < 512; kt += 32) {
      v16bf af = load_a_frag(abase, Nres, kt);
      v16bf bf;
      if (ct < 16) bf = load_b_frag_sw(vb, (kt >> 5) * 16 + ct);
      else         bf = load_b_frag_sw(vpb, (kt >> 5) * 3 + (ct - 16));
      acc = wmma_bf16(af, bf, acc);
    }
    int c = col0 + nlo;
#pragma unroll
    for (int v = 0; v < 8; ++v) {
      int i = m0 + wave * 16 + mhi + v;
      if (c < 256)      feats[(size_t)i * FEAT + h * 256 + c] = acc[v];
      else if (c < 292) optg[((size_t)h * Nres + i) * 36 + (c - 256)] = acc[v];
    }
  }
}

// ---------------------------------------------------------------------------
// o_pair[i,h,d] = sum_j a[h,i,j] * pair_z[i,j,d]  (M=8 heads padded to 16).
// ---------------------------------------------------------------------------
__global__ __launch_bounds__(32) void opair_kernel(const __bf16* __restrict__ a_bf,
                                                   const __bf16* __restrict__ pairz,
                                                   float* __restrict__ feats) {
  const int i = blockIdx.x;
  const int l = lane_id(), nlo = l & 15, mhi = (l < 16) ? 0 : 8;
  const int m = l & 15;
  const int khalf = (l < 16) ? 0 : 8;
  const u32* pzb = (const u32*)pairz + (size_t)i * 8192;
  v8f acc[2] = {};
  for (int kt = 0; kt < 512; kt += 32) {
    u32x8 w = {};
    if (m < 8) {
      const u32x4* p =
          (const u32x4*)(a_bf + ((size_t)m * Nres + i) * Nres + kt + khalf);
      u32x4 lo = p[0];
      u32x4 hi = p[2];
      w[0] = lo[0]; w[1] = lo[1]; w[2] = lo[2]; w[3] = lo[3];
      w[4] = hi[0]; w[5] = hi[1]; w[6] = hi[2]; w[7] = hi[3];
    }
    v16bf af = __builtin_bit_cast(v16bf, w);
#pragma unroll
    for (int n = 0; n < 2; ++n) {
      v16bf bf = load_b_frag_sw(pzb, (kt >> 5) * 2 + n);
      acc[n] = wmma_bf16(af, bf, acc[n]);
    }
  }
#pragma unroll
  for (int n = 0; n < 2; ++n) {
#pragma unroll
    for (int v = 0; v < 8; ++v) {
      int mr = mhi + v;
      if (mr < 8)
        feats[(size_t)i * FEAT + 2432 + mr * 32 + n * 16 + nlo] = acc[n][v];
    }
  }
}

// ---------------------------------------------------------------------------
// Inverse rigid transform of o_pt + norms into feats.
// ---------------------------------------------------------------------------
__global__ void optfin_kernel(const float* __restrict__ optg,
                              const float* __restrict__ rot,
                              const float* __restrict__ trans,
                              float* __restrict__ feats) {
  int idx = blockIdx.x * blockDim.x + threadIdx.x;
  if (idx >= Hh * Nres * PVp) return;
  int p = idx % PVp;
  int i = (idx / PVp) % Nres;
  int h = idx / (PVp * Nres);
  const float* R = rot + i * 9;
  const float* T = trans + i * 3;
  const float* o = optg + ((size_t)h * Nres + i) * 36 + p * 3;
  float vx = o[0] - T[0], vy = o[1] - T[1], vz = o[2] - T[2];
  float x = R[0] * vx + R[3] * vy + R[6] * vz;   // R^T * v
  float y = R[1] * vx + R[4] * vy + R[7] * vz;
  float z = R[2] * vx + R[5] * vy + R[8] * vz;
  float nrm = sqrtf(x * x + y * y + z * z + 1e-8f);
  size_t base = (size_t)i * FEAT + 2048;
  int c = h * PVp + p;
  feats[base + c]       = x;
  feats[base + 96 + c]  = y;
  feats[base + 192 + c] = z;
  feats[base + 288 + c] = nrm;
}

// ---------------------------------------------------------------------------
extern "C" void kernel_launch(void* const* d_in, const int* in_sizes, int n_in,
                              void* d_out, int out_size, void* d_ws, size_t ws_size,
                              hipStream_t stream) {
  (void)in_sizes; (void)n_in; (void)out_size; (void)ws_size;
  const float* s     = (const float*)d_in[0];
  const float* z     = (const float*)d_in[1];
  const float* rot   = (const float*)d_in[2];
  const float* trans = (const float*)d_in[3];
  const float* mask  = (const float*)d_in[4];
  const float* wq    = (const float*)d_in[5];
  const float* bq    = (const float*)d_in[6];
  const float* wkv   = (const float*)d_in[7];
  const float* bkv   = (const float*)d_in[8];
  const float* wqp   = (const float*)d_in[9];
  const float* bqp   = (const float*)d_in[10];
  const float* wkvp  = (const float*)d_in[11];
  const float* bkvp  = (const float*)d_in[12];
  const float* wb    = (const float*)d_in[13];
  const float* bb    = (const float*)d_in[14];
  const float* wz    = (const float*)d_in[15];
  const float* bz    = (const float*)d_in[16];
  const float* hwgt  = (const float*)d_in[17];
  const float* wo    = (const float*)d_in[18];
  const float* bo    = (const float*)d_in[19];
  float* out = (float*)d_out;

  char* ws = (char*)d_ws;
  size_t off = 0;
  auto alloc = [&](size_t bytes) -> void* {
    void* p = ws + off;
    off = (off + bytes + 255) & ~(size_t)255;
    return p;
  };
  float*  qbuf   = (float*)alloc((size_t)Nres * 2048 * 4);
  float*  kvbuf  = (float*)alloc((size_t)Nres * 4096 * 4);
  float*  qpbuf  = (float*)alloc((size_t)Nres * 192 * 4);
  float*  kvpbuf = (float*)alloc((size_t)Nres * 480 * 4);
  __bf16* qcat   = (__bf16*)alloc((size_t)Hh * Nres * KCAT * 2);
  __bf16* kcatT  = (__bf16*)alloc((size_t)Hh * KCAT * Nres * 2);
  __bf16* vbf    = (__bf16*)alloc((size_t)Hh * Nres * 256 * 2);
  __bf16* vptbf  = (__bf16*)alloc((size_t)Hh * Nres * 48 * 2);
  float*  Qsq    = (float*)alloc((size_t)Nres * 8 * 4);
  float*  Ksq    = (float*)alloc((size_t)Nres * 8 * 4);
  float*  bbias  = (float*)alloc((size_t)Hh * NM * 4);
  __bf16* pairz  = (__bf16*)alloc((size_t)NM * 32 * 2);
  __bf16* a_bf   = (__bf16*)alloc((size_t)Hh * Nres * Nres * 2);
  float*  optg   = (float*)alloc((size_t)Hh * Nres * 36 * 4);
  float*  feats  = (float*)alloc((size_t)Nres * FEAT * 4);

  // s projections
  gemm_f32_bf16<<<dim3(2048 / 64, Nres / 64), 128, 0, stream>>>(s, wq, bq, qbuf,
                                                                Nres, 2048, CSd);
  gemm_f32_bf16<<<dim3(4096 / 64, Nres / 64), 128, 0, stream>>>(s, wkv, bkv, kvbuf,
                                                                Nres, 4096, CSd);
  gemm_f32_bf16<<<dim3(3, Nres / 64), 128, 0, stream>>>(s, wqp, bqp, qpbuf, Nres,
                                                        192, CSd);
  gemm_f32_bf16<<<dim3(8, Nres / 64), 128, 0, stream>>>(s, wkvp, bkvp, kvpbuf, Nres,
                                                        480, CSd);
  // fused z projections (single pass over the 134MB z tensor)
  zproj_kernel<<<NM / 64, 128, 0, stream>>>(z, wb, bb, wz, bz, bbias, pairz);
  // rigid transforms + operand packing
  pack_kernel<<<Nres, 256, 0, stream>>>(qbuf, kvbuf, qpbuf, kvpbuf, rot, trans,
                                        hwgt, qcat, kcatT, vbf, vptbf, Qsq, Ksq);
  // attention logits + softmax
  logits_softmax_kernel<<<dim3(Nres / 16, Hh), 128, 0, stream>>>(
      qcat, kcatT, bbias, Qsq, Ksq, mask, hwgt, a_bf);
  // a@v and a@v_pts
  av_kernel<<<dim3(Nres / 64, Hh), 128, 0, stream>>>(a_bf, vbf, vptbf, feats, optg);
  // o_pair
  opair_kernel<<<Nres, 32, 0, stream>>>(a_bf, pairz, feats);
  // inverse rigid + norms
  optfin_kernel<<<(Hh * Nres * PVp + 255) / 256, 256, 0, stream>>>(optg, rot, trans,
                                                                   feats);
  // output projection
  gemm_f32_bf16<<<dim3(CSd / 64, Nres / 64), 128, 0, stream>>>(feats, wo, bo, out,
                                                               Nres, CSd, FEAT);
}
